// BiLSTM_CRF_71124658421884
// MI455X (gfx1250) — matrix operand order repaired and checked by reading, not compiled, not run on hardware
//
#include <hip/hip_runtime.h>

// ---------------------------------------------------------------------------
// BiLSTM-CRF for MI455X (gfx1250, wave32).
//   K1: f32 -> bf16 conversion (x, w_ih x2, w_hh x2)
//   K2: input projection GEMM: v_wmma_f32_16x16x32_bf16, B tiles staged into
//       LDS by the Tensor Data Mover (tensor_load_to_lds + s_wait_tensorcnt),
//       double-buffered; bias (b_ih + b_hh) folded into the output.
//   K3: persistent-grid LSTM recurrence (64 WGs, software step barrier)
//   K4: tag projection (wave reduction)
//   K5: Viterbi (single thread, 4 tags)
// ---------------------------------------------------------------------------

typedef __attribute__((ext_vector_type(16))) __bf16 v16bf;
typedef __attribute__((ext_vector_type(8)))  __bf16 v8bf;
typedef __attribute__((ext_vector_type(8)))  float  v8f;
typedef __attribute__((ext_vector_type(4)))  unsigned int u32x4;
typedef __attribute__((ext_vector_type(8)))  int    i32x8;
typedef __attribute__((ext_vector_type(4)))  int    i32x4;

#define T_LEN 4096
#define F_DIM 1024
#define H2    1024
#define NGATE 4096   // 4*H2
#define HID   2048   // 2*H2

#if defined(__has_builtin)
#if __has_builtin(__builtin_amdgcn_tensor_load_to_lds) && \
    __has_builtin(__builtin_amdgcn_s_wait_tensorcnt)
#define HAVE_TDM 1
#endif
#endif
#ifndef HAVE_TDM
#define HAVE_TDM 0
#endif

__device__ __forceinline__ unsigned short f2bf(float f) {
    unsigned u = __float_as_uint(f);
    unsigned r = u + 0x7FFFu + ((u >> 16) & 1u);   // round-to-nearest-even
    return (unsigned short)(r >> 16);
}
__device__ __forceinline__ float sigmoidf_(float x) {
    return 1.0f / (1.0f + __expf(-x));
}

// ---------------------------------------------------------------- K1: convert
__global__ void f32_to_bf16_kernel(const float* __restrict__ src,
                                   unsigned short* __restrict__ dst, int n) {
    for (int i = blockIdx.x * blockDim.x + threadIdx.x; i < n;
         i += gridDim.x * blockDim.x)
        dst[i] = f2bf(src[i]);
}

#if HAVE_TDM
// TDM descriptor (ISA 8.3/8.4): 2D tile, 32 elems (contiguous K) x 64 rows,
// 2B elements, row stride 1024 elems; dest = packed 64x32x2B (4KB) in LDS.
// workgroup_mask = 0 (not in a cluster).
__device__ __forceinline__ void tdm_load_tile_b(const unsigned short* gsrc,
                                                unsigned lds_byte_addr) {
    unsigned long long ga = (unsigned long long)(size_t)gsrc;
    u32x4 g0;
    g0[0] = 1u;                                    // count=1, user mode
    g0[1] = lds_byte_addr;                         // lds_addr
    g0[2] = (unsigned)(ga & 0xFFFFFFFFu);          // global_addr[31:0]
    g0[3] = (unsigned)((ga >> 32) & 0x01FFFFFFu)   // global_addr[56:32]
            | (2u << 30);                          // type = 2 ("image")
    i32x8 g1;
    g1[0] = 0x00010000;                            // data_size=1 (2B), mask=0
    g1[1] = (int)((F_DIM & 0xFFFFu) << 16);        // tensor_dim0[15:0]
    g1[2] = (int)(((unsigned)F_DIM >> 16) |
                  ((NGATE & 0xFFFFu) << 16));      // tensor_dim0 hi | dim1 lo
    g1[3] = (int)(((unsigned)NGATE >> 16) |
                  (32u << 16));                    // tensor_dim1 hi | tile_dim0
    g1[4] = 64;                                    // tile_dim1=64, tile_dim2=0
    g1[5] = F_DIM;                                 // tensor_dim0_stride[31:0]
    g1[6] = 0;                                     // stride hi | dim1_stride lo
    g1[7] = 0;
    i32x4 z4 = {0, 0, 0, 0};
#if __clang_major__ >= 23
    i32x8 z8 = {0, 0, 0, 0, 0, 0, 0, 0};
    __builtin_amdgcn_tensor_load_to_lds(g0, g1, z4, z4, z8, 0);
#else
    __builtin_amdgcn_tensor_load_to_lds(g0, g1, z4, z4, 0);
#endif
}
#endif

// ------------------------------------------------- K2: projection GEMM (WMMA)
// xp[dir][s][n] = bf16( x[row(s)] ) . bf16( w_ih[dir][n] ) + b_ih[n] + b_hh[n]
// row(s) = s (fwd) or T-1-s (bwd).  Block: 8 waves = 128(M) x 64(N).
// B tile (64 rows x 32 K, bf16) is shared by all 8 waves -> staged in LDS
// via the Tensor Data Mover, double-buffered.
__global__ __launch_bounds__(256)
void proj_wmma_kernel(const unsigned short* __restrict__ xb,     // [T,F] bf16
                      const unsigned short* __restrict__ wihb,   // [2][NGATE,F]
                      const float* __restrict__ b_ih_f, const float* __restrict__ b_hh_f,
                      const float* __restrict__ b_ih_b, const float* __restrict__ b_hh_b,
                      float* __restrict__ xp)                    // [2][T,NGATE]
{
    __shared__ __attribute__((aligned(32))) unsigned short bsh[2][64 * 32];

    const int dir  = blockIdx.z;
    const int lane = threadIdx.x & 31;
    const int w    = threadIdx.x >> 5;
    const int r16  = lane & 15;
    const int hsel = lane >> 4;              // 0: lanes 0-15, 1: lanes 16-31
    const int mt   = blockIdx.y * 128 + w * 16;
    const int nt   = blockIdx.x * 64;

    const unsigned short* wptr = wihb + (size_t)dir * NGATE * F_DIM
                                      + (size_t)nt * F_DIM;   // B tile row base

    v8f acc[4];
#pragma unroll
    for (int j = 0; j < 4; ++j)
#pragma unroll
        for (int v = 0; v < 8; ++v) acc[j][v] = 0.0f;

    int arow = mt + r16;
    if (dir) arow = T_LEN - 1 - arow;
    const unsigned short* abase = xb + (size_t)arow * F_DIM;

#if HAVE_TDM
    if (w == 0)                               // preload tile 0 (k=0)
        tdm_load_tile_b(wptr, (unsigned)(size_t)&bsh[0][0]);
#endif

    for (int ki = 0; ki < F_DIM / 32; ++ki) {
        const int k   = ki * 32;
        const int cur = ki & 1;

#if HAVE_TDM
        if (w == 0) {
            if (ki + 1 < F_DIM / 32) {        // issue next, wait for current
                tdm_load_tile_b(wptr + (k + 32),
                                (unsigned)(size_t)&bsh[cur ^ 1][0]);
                __builtin_amdgcn_s_wait_tensorcnt(1);
            } else {
                __builtin_amdgcn_s_wait_tensorcnt(0);
            }
        }
        __syncthreads();                      // tile `cur` visible to all waves
#else
        {   // cooperative fallback: 256 threads x 16B = 4KB tile
            const int row = threadIdx.x >> 2, seg = threadIdx.x & 3;
            *(uint4*)&bsh[cur][row * 32 + seg * 8] =
                *(const uint4*)(wptr + (size_t)row * F_DIM + k + seg * 8);
        }
        __syncthreads();
#endif

        // A fragment (16x32 bf16): lane<16 holds K={k..k+15},
        // lane>=16 holds K={k+16..k+31}.
        const int k0 = k + hsel * 16;
        v8bf alo = *(const v8bf*)(abase + k0);
        v8bf ahi = *(const v8bf*)(abase + k0 + 8);
        v16bf a;
#pragma unroll
        for (int i = 0; i < 8; ++i) { a[i] = alo[i]; a[8 + i] = ahi[i]; }

        // B fragments (32x16 bf16) from LDS: lane holds column l%16,
        // contiguous K = 16*hsel .. +15 within the staged tile.
#pragma unroll
        for (int j = 0; j < 4; ++j) {
            v16bf b = *(const v16bf*)&bsh[cur][(j * 16 + r16) * 32 + hsel * 16];
            acc[j] = __builtin_amdgcn_wmma_f32_16x16x32_bf16(
                false, a, false, b, (short)0, acc[j], false, false);
        }
        __syncthreads();                      // release tile before refill
    }

    const float* bih = dir ? b_ih_b : b_ih_f;
    const float* bhh = dir ? b_hh_b : b_hh_f;
    float* xpd = xp + (size_t)dir * T_LEN * NGATE;
#pragma unroll
    for (int j = 0; j < 4; ++j) {
        const int n = nt + j * 16 + r16;
        const float bias = bih[n] + bhh[n];
#pragma unroll
        for (int v = 0; v < 8; ++v) {
            const int m = mt + v + hsel * 8;          // D: row = vgpr + 8*(l/16)
            xpd[(size_t)m * NGATE + n] = acc[j][v] + bias;
        }
    }
}

// ------------------------------------------------- K3: persistent recurrence
// 64 WGs: dir = blk/32, slice = blk%32 -> hidden units [32*slice, +32).
// Gate rows owned: {g*1024 + j0 + u : g in 0..3, u in 0..31} (128 rows x K=1024).
// h history lives in lstm_out; previous-step h is read back through L2.
__global__ __launch_bounds__(256)
void lstm_rec_kernel(const unsigned short* __restrict__ whhb,  // [2][NGATE,H2] bf16
                     const float* __restrict__ xp,             // [2][T,NGATE]
                     const float* __restrict__ h0,             // [2][H2]
                     const float* __restrict__ c0,             // [2][H2]
                     float* __restrict__ lstm_out,             // [T,HID]
                     int* __restrict__ cnt)                    // [2][T]
{
    __shared__ float hsh[H2];
    __shared__ float gates[128];
    __shared__ float cst[32];

    const int tid  = threadIdx.x;
    const int lane = tid & 31;
    const int w    = tid >> 5;
    const int dir  = blockIdx.x >> 5;
    const int j0   = (blockIdx.x & 31) * 32;

    const unsigned short* wbase = whhb + (size_t)dir * NGATE * H2;
    const float* xpd = xp + (size_t)dir * T_LEN * NGATE;
    int* mycnt = cnt + dir * T_LEN;

    if (tid < 32) cst[tid] = c0[dir * H2 + j0 + tid];

    for (int s = 0; s < T_LEN; ++s) {
        if (s > 0) {                       // wait for all 32 WGs of this dir
            if (tid == 0) {
                while (__hip_atomic_load(&mycnt[s - 1], __ATOMIC_ACQUIRE,
                                         __HIP_MEMORY_SCOPE_AGENT) < 32)
                    __builtin_amdgcn_s_sleep(1);
            }
            __syncthreads();
        }
        // stage h_{s-1} (or h0) into LDS
        const float* hsrc = (s == 0)
            ? (h0 + dir * H2)
            : (lstm_out + (size_t)(dir == 0 ? s - 1 : T_LEN - s) * HID + dir * H2);
        for (int i = tid; i < H2; i += 256) hsh[i] = hsrc[i];
        __syncthreads();

        // each wave: 16 gate rows, coalesced b128 bf16 weight loads
#pragma unroll 1
        for (int rr = 0; rr < 16; ++rr) {
            const int lr   = w * 16 + rr;
            const int grow = (lr >> 5) * H2 + j0 + (lr & 31);
            const unsigned short* wr = wbase + (size_t)grow * H2;
            if (rr + 1 < 16) {             // hide L2->L0 latency of next row
                const int nrow = ((lr + 1) >> 5) * H2 + j0 + ((lr + 1) & 31);
                __builtin_prefetch(wbase + (size_t)nrow * H2 + lane * 8, 0, 1);
            }
            float p = 0.0f;
#pragma unroll
            for (int c4 = 0; c4 < 4; ++c4) {
                const int kk = c4 * 256 + lane * 8;
                uint4 raw = *(const uint4*)(wr + kk);
                const unsigned* rw = (const unsigned*)&raw;
#pragma unroll
                for (int q = 0; q < 4; ++q) {
                    float w0 = __uint_as_float(rw[q] << 16);
                    float w1 = __uint_as_float(rw[q] & 0xFFFF0000u);
                    p = fmaf(w0, hsh[kk + 2 * q + 0], p);
                    p = fmaf(w1, hsh[kk + 2 * q + 1], p);
                }
            }
#pragma unroll
            for (int off = 16; off > 0; off >>= 1) p += __shfl_xor(p, off, 32);
            if (lane == 0) gates[lr] = p + xpd[(size_t)s * NGATE + grow];
        }
        __syncthreads();

        if (tid < 32) {                    // gate order i,f,g,o
            const int u = tid;
            float gi = gates[u], gf = gates[32 + u];
            float gg = gates[64 + u], go = gates[96 + u];
            float c = sigmoidf_(gf) * cst[u] + sigmoidf_(gi) * tanhf(gg);
            float h = sigmoidf_(go) * tanhf(c);
            cst[u] = c;
            const int time = (dir == 0) ? s : (T_LEN - 1 - s);
            lstm_out[(size_t)time * HID + dir * H2 + j0 + u] = h;
        }
        __threadfence();
        __syncthreads();
        if (tid == 0)
            __hip_atomic_fetch_add(&mycnt[s], 1, __ATOMIC_RELEASE,
                                   __HIP_MEMORY_SCOPE_AGENT);
    }
}

// ------------------------------------------------------- K4: tag projection
__global__ __launch_bounds__(256)
void feats_kernel(const float* __restrict__ lstm_out,
                  const float* __restrict__ w_tag,   // [4,HID]
                  const float* __restrict__ b_tag,   // [4]
                  float* __restrict__ feats)         // [T,4]
{
    const int lane = threadIdx.x & 31;
    const int t = blockIdx.x * 8 + (threadIdx.x >> 5);
    const float* hrow = lstm_out + (size_t)t * HID;
    float p0 = 0, p1 = 0, p2 = 0, p3 = 0;
    for (int j = lane; j < HID; j += 32) {
        float h = hrow[j];
        p0 = fmaf(h, w_tag[j], p0);
        p1 = fmaf(h, w_tag[HID + j], p1);
        p2 = fmaf(h, w_tag[2 * HID + j], p2);
        p3 = fmaf(h, w_tag[3 * HID + j], p3);
    }
#pragma unroll
    for (int off = 16; off > 0; off >>= 1) {
        p0 += __shfl_xor(p0, off, 32);
        p1 += __shfl_xor(p1, off, 32);
        p2 += __shfl_xor(p2, off, 32);
        p3 += __shfl_xor(p3, off, 32);
    }
    if (lane == 0) {
        feats[t * 4 + 0] = p0 + b_tag[0];
        feats[t * 4 + 1] = p1 + b_tag[1];
        feats[t * 4 + 2] = p2 + b_tag[2];
        feats[t * 4 + 3] = p3 + b_tag[3];
    }
}

// ------------------------------------------------------------- K5: Viterbi
// transitions[next, prev]; START=2, STOP=3; first-max tie-break (jnp.argmax).
__global__ void viterbi_kernel(const float* __restrict__ feats,
                               const float* __restrict__ trans,
                               int* __restrict__ bp,          // [T,4]
                               float* __restrict__ out)       // [1 + T]
{
    if (threadIdx.x != 0 || blockIdx.x != 0) return;
    float fv[4];
    for (int n = 0; n < 4; ++n) fv[n] = (n == 2) ? 0.0f : -10000.0f;
    for (int t = 0; t < T_LEN; ++t) {
        float nf[4];
        for (int nx = 0; nx < 4; ++nx) {
            float best = -3.4e38f; int bi = 0;
            for (int pv = 0; pv < 4; ++pv) {
                float sc = fv[pv] + trans[nx * 4 + pv];
                if (sc > best) { best = sc; bi = pv; }
            }
            bp[t * 4 + nx] = bi;
            nf[nx] = best + feats[t * 4 + nx];
        }
        for (int n = 0; n < 4; ++n) fv[n] = nf[n];
    }
    float best = -3.4e38f; int bt = 0;
    for (int n = 0; n < 4; ++n) {
        float sc = fv[n] + trans[3 * 4 + n];
        if (sc > best) { best = sc; bt = n; }
    }
    out[0] = best;
    int tag = bt;
    for (int t = T_LEN - 1; t >= 0; --t) { out[1 + t] = (float)tag; tag = bp[t * 4 + tag]; }
}

// ---------------------------------------------------------------------------
extern "C" void kernel_launch(void* const* d_in, const int* in_sizes, int n_in,
                              void* d_out, int out_size, void* d_ws, size_t ws_size,
                              hipStream_t stream) {
    (void)in_sizes; (void)n_in; (void)out_size; (void)ws_size;
    const float* sentence = (const float*)d_in[0];
    const float* h0       = (const float*)d_in[1];
    const float* c0       = (const float*)d_in[2];
    const float* w_ih_f   = (const float*)d_in[3];
    const float* w_hh_f   = (const float*)d_in[4];
    const float* b_ih_f   = (const float*)d_in[5];
    const float* b_hh_f   = (const float*)d_in[6];
    const float* w_ih_b   = (const float*)d_in[7];
    const float* w_hh_b   = (const float*)d_in[8];
    const float* b_ih_b   = (const float*)d_in[9];
    const float* b_hh_b   = (const float*)d_in[10];
    const float* w_tag    = (const float*)d_in[11];
    const float* b_tag    = (const float*)d_in[12];
    const float* trans    = (const float*)d_in[13];
    float* out = (float*)d_out;

    // ---- workspace carve (256B aligned) ----
    char* base = (char*)d_ws;
    size_t off = 0;
    auto carve = [&](size_t bytes) -> char* {
        char* p = base + off;
        off = (off + bytes + 255) & ~(size_t)255;
        return p;
    };
    unsigned short* xb    = (unsigned short*)carve((size_t)T_LEN * F_DIM * 2);
    unsigned short* wihb  = (unsigned short*)carve((size_t)2 * NGATE * F_DIM * 2);
    unsigned short* whhb  = (unsigned short*)carve((size_t)2 * NGATE * H2 * 2);
    float* xp             = (float*)carve((size_t)2 * T_LEN * NGATE * 4);
    float* lstm_out       = (float*)carve((size_t)T_LEN * HID * 4);
    float* feats          = (float*)carve((size_t)T_LEN * 4 * 4);
    int*   bp             = (int*)carve((size_t)T_LEN * 4 * 4);
    int*   cnt            = (int*)carve((size_t)2 * T_LEN * 4);

    // step-barrier counters must be zero each call (graph-replay safe)
    hipMemsetAsync(cnt, 0, (size_t)2 * T_LEN * 4, stream);

    // ---- K1: bf16 conversions ----
    const int CB = 256, CG = 2048;
    f32_to_bf16_kernel<<<CG, CB, 0, stream>>>(sentence, xb, T_LEN * F_DIM);
    f32_to_bf16_kernel<<<CG, CB, 0, stream>>>(w_ih_f, wihb, NGATE * F_DIM);
    f32_to_bf16_kernel<<<CG, CB, 0, stream>>>(w_ih_b, wihb + (size_t)NGATE * F_DIM, NGATE * F_DIM);
    f32_to_bf16_kernel<<<CG, CB, 0, stream>>>(w_hh_f, whhb, NGATE * H2);
    f32_to_bf16_kernel<<<CG, CB, 0, stream>>>(w_hh_b, whhb + (size_t)NGATE * H2, NGATE * H2);

    // ---- K2: input projection GEMM (both directions) ----
    dim3 pgrid(NGATE / 64, T_LEN / 128, 2);
    proj_wmma_kernel<<<pgrid, 256, 0, stream>>>(xb, wihb, b_ih_f, b_hh_f,
                                                b_ih_b, b_hh_b, xp);

    // ---- K3: persistent recurrence (64 co-resident WGs) ----
    lstm_rec_kernel<<<64, 256, 0, stream>>>(whhb, xp, h0, c0, lstm_out, cnt);

    // ---- K4: tag projection ----
    feats_kernel<<<T_LEN / 8, 256, 0, stream>>>(lstm_out, w_tag, b_tag, feats);

    // ---- K5: Viterbi decode ----
    viterbi_kernel<<<1, 32, 0, stream>>>(feats, trans, bp, out);
}
